// ResidualRecurrentEncoder_82411832476036
// MI455X (gfx1250) — matrix operand order repaired and checked by reading, not compiled, not run on hardware
//
#include <hip/hip_runtime.h>
#include <hip/hip_bf16.h>

// ---------------------------------------------------------------------------
// Residual bidirectional LSTM encoder for MI455X (gfx1250, wave32, WMMA).
//   * all matmuls in bf16 WMMA (v_wmma_f32_16x16x32_bf16), f32 accumulate
//   * per layer: batched input-projection GEMM -> xg stored bf16, transposed
//     per timestep as [T][4H][B] so every fragment access is a 16B vector
//   * persistent 1-workgroup scan: w_hh (128KB) + h double-buffer (32KB) +
//     xg slab double-buffer (128KB) all LDS-resident (288KB of 320KB/WGP);
//     next timestep's xg slab is fetched with GLOBAL_LOAD_ASYNC_TO_LDS_B128
//     (ASYNCcnt) while the current step computes.
// ---------------------------------------------------------------------------

#define Tt   512
#define Bb   64
#define Hh   128
#define G4   512      // 4*H
#define TBr  (Tt*Bb)  // 32768 rows
#define SLAB (Bb*G4)  // bf16 elements per timestep slab (64KB)

typedef __attribute__((ext_vector_type(16))) __bf16 v16bf;
typedef __attribute__((ext_vector_type(8)))  __bf16 v8bf;
typedef __attribute__((ext_vector_type(8)))  float  v8f;
typedef __attribute__((ext_vector_type(4)))  int    v4i;

#if defined(__gfx1250__) && \
    __has_builtin(__builtin_amdgcn_global_load_async_to_lds_b128) && \
    __has_builtin(__builtin_amdgcn_s_wait_asynccnt)
#define HAVE_ASYNC_LDS 1
#define AS1 __attribute__((address_space(1)))
#define AS3 __attribute__((address_space(3)))
#endif

__device__ __forceinline__ __bf16 f2bf(float f) {
  union { float f; unsigned u; } in; in.f = f;
  unsigned r = in.u + 0x7FFFu + ((in.u >> 16) & 1u);   // round-to-nearest-even
  union { unsigned short s; __bf16 b; } out;
  out.s = (unsigned short)(r >> 16);
  return out.b;
}

__device__ __forceinline__ float bf2f(__bf16 b) {
  union { unsigned short s; __bf16 b; } in; in.b = b;
  union { unsigned u; float f; } out; out.u = ((unsigned)in.s) << 16;
  return out.f;
}

__device__ __forceinline__ float sigf(float x) { return 1.0f / (1.0f + expf(-x)); }

// f32 -> bf16 elementwise (weights)
__global__ void cvt_bf16_kernel(const float* __restrict__ in,
                                __bf16* __restrict__ out, int n) {
  int i = blockIdx.x * 256 + threadIdx.x;
  if (i < n) out[i] = f2bf(in[i]);
}

// embedding gather + bf16 convert
__global__ void embed_kernel(const int* __restrict__ ids,
                             const float* __restrict__ emb,
                             __bf16* __restrict__ out, int total) {
  int i = blockIdx.x * 256 + threadIdx.x;
  if (i >= total) return;
  int row = i >> 7, col = i & 127;
  out[i] = f2bf(emb[(size_t)ids[row] * Hh + col]);
}

// ---------------------------------------------------------------------------
// Input projection: xgT[t][n][m] = bf16( X[t*B+m,:K] . W[n,:K] + bias[n] )
// One 16x16 output tile per wave; per-t 16-row tiles never straddle t (64%16==0).
// ---------------------------------------------------------------------------
__global__ __launch_bounds__(256)
void gemm_xg_kernel(const __bf16* __restrict__ X,   // [rows, K] row-major
                    const __bf16* __restrict__ W,   // [512, K]  row-major (B = W^T)
                    const float*  __restrict__ bias,// [512]
                    __bf16* __restrict__ out,       // [T, 512, 64] bf16 (transposed)
                    int K) {
  const int wid  = threadIdx.x >> 5;
  const int lane = threadIdx.x & 31;
  const int tile = blockIdx.x * 8 + wid;
  const int tiles_n = G4 / 16;                      // 32
  const int m0 = (tile / tiles_n) * 16;             // row = t*64 + batch
  const int n0 = (tile % tiles_n) * 16;
  const int l16 = lane & 15, lhi = lane >> 4;

  v8f acc = {};
  const __bf16* xrow = X + (size_t)(m0 + l16) * K + lhi * 8;   // A: K 0-7/8-15 split
  const __bf16* wrow = W + (size_t)(n0 + l16) * K + lhi * 8;   // B: K 0-7/8-15 split

  for (int k0 = 0; k0 < K; k0 += 32) {
    v8bf alo = *(const v8bf*)(xrow + k0);
    v8bf ahi = *(const v8bf*)(xrow + k0 + 16);
    v8bf blo = *(const v8bf*)(wrow + k0);
    v8bf bhi = *(const v8bf*)(wrow + k0 + 16);
    v16bf a, b;
#pragma unroll
    for (int q = 0; q < 8; ++q) { a[q] = alo[q]; a[q + 8] = ahi[q];
                                  b[q] = blo[q]; b[q + 8] = bhi[q]; }
    acc = __builtin_amdgcn_wmma_f32_16x16x32_bf16(false, a, false, b,
                                                  (short)0, acc, false, false);
  }
  const float bv = bias[n0 + l16];
  // transposed store: lane owns (n0+l16, m = m0 + lhi*8 + v), v contiguous -> b128
  const int t = m0 >> 6;
  __bf16* op = out + ((size_t)t * G4 + n0 + l16) * Bb + (m0 & 63) + lhi * 8;
  v8bf ov;
#pragma unroll
  for (int v = 0; v < 8; ++v) ov[v] = f2bf(acc[v] + bv);
  *(v8bf*)op = ov;
}

// ---------------------------------------------------------------------------
// Persistent LSTM scan: single workgroup, 32 waves (4 row-tiles x 8 col-tiles).
// LDS: whh bf16 [512x128] (128KB) | h bf16 [2][64][128] (32KB)
//      | xg slab bf16 [2][512][64] (128KB, async double-buffered)
// ---------------------------------------------------------------------------
__global__ __launch_bounds__(1024)
void lstm_scan_kernel(const __bf16* __restrict__ xg,  // [T,4H,B] bf16 (transposed)
                      const __bf16* __restrict__ whh, // [4H,H] row-major
                      float*  __restrict__ out_f, int out_stride, int col_off,
                      __bf16* __restrict__ out_bf, int bf_stride, int bf_off,
                      const float* __restrict__ resid, // [T,B,H] or null
                      float* __restrict__ hn, float* __restrict__ cn,
                      int reverse) {
  extern __shared__ char smem[];
  __bf16* whh_s = (__bf16*)smem;                               // 512*128
  __bf16* hbuf  = (__bf16*)(smem + G4 * Hh * 2);               // [2][64][128]
  __bf16* xgbuf = (__bf16*)(smem + G4 * Hh * 2 + 2 * Bb * Hh * 2); // [2][512][64]

  const int tid = threadIdx.x;
  { // cooperative: stage whh, zero h buffers, preload first xg slab
    const unsigned* src = (const unsigned*)whh;
    unsigned* dst = (unsigned*)whh_s;
    for (int i = tid; i < G4 * Hh / 2; i += 1024) dst[i] = src[i];
    unsigned* hz = (unsigned*)hbuf;
    for (int i = tid; i < 2 * Bb * Hh / 2; i += 1024) hz[i] = 0u;
    const int tt0 = reverse ? (Tt - 1) : 0;
    const uint4* s4 = (const uint4*)(xg + (size_t)tt0 * SLAB);
    uint4* d4 = (uint4*)xgbuf;
    for (int i = tid; i < SLAB * 2 / 16; i += 1024) d4[i] = s4[i];
  }
  __syncthreads();

  const int wid = tid >> 5, lane = tid & 31;
  const int r   = wid >> 3;                 // batch row-tile 0..3
  const int nh  = wid & 7;                  // hidden col-tile 0..7
  const int l16 = lane & 15, lhi = lane >> 4;
  const int j   = nh * 16 + l16;            // hidden column owned in C/D frags

  float cstate[8];
#pragma unroll
  for (int v = 0; v < 8; ++v) cstate[v] = 0.0f;

  int p = 0;
  for (int s = 0; s < Tt; ++s) {
    const int tt = reverse ? (Tt - 1 - s) : s;

    // --- issue async copy of next timestep's xg slab into buffer p^1
    if (s + 1 < Tt) {
      const int tn = reverse ? (Tt - 2 - s) : (s + 1);
      const __bf16* gsrc = xg + (size_t)tn * SLAB;
      __bf16* ldst = xgbuf + (p ^ 1) * SLAB;
#ifdef HAVE_ASYNC_LDS
#pragma unroll
      for (int q = 0; q < 4; ++q) {
        const int idx = (tid * 4 + q) * 8;           // 8 bf16 = 16B chunks
        __builtin_amdgcn_global_load_async_to_lds_b128(
            (AS1 v4i*)(gsrc + idx), (AS3 v4i*)(ldst + idx), 0, 0);
      }
#else
      const uint4* s4 = (const uint4*)gsrc;
      uint4* d4 = (uint4*)ldst;
#pragma unroll
      for (int q = 0; q < 4; ++q) d4[tid * 4 + q] = s4[tid * 4 + q];
#endif
      if (resid)  // prefetch next residual slab (global_prefetch_b8)
        __builtin_prefetch(resid + (size_t)tn * Bb * Hh + tid * 8, 0, 0);
    }

    // --- A fragments (h rows 16r..16r+15) from LDS, 4 K-chunks of 32
    v16bf afrag[4];
    {
      const __bf16* hrow = hbuf + p * Bb * Hh + (r * 16 + l16) * Hh + lhi * 8;
#pragma unroll
      for (int kk = 0; kk < 4; ++kk) {
        v8bf lo = *(const v8bf*)(hrow + kk * 32);
        v8bf hi = *(const v8bf*)(hrow + kk * 32 + 16);
        v16bf a;
#pragma unroll
        for (int q = 0; q < 8; ++q) { a[q] = lo[q]; a[q + 8] = hi[q]; }
        afrag[kk] = a;
      }
    }

    // --- 4 gate tiles: init from LDS xg slab (one b128 each), WMMA accumulate
    float gv[4][8];
    const __bf16* xgs = xgbuf + p * SLAB;
#pragma unroll
    for (int gi = 0; gi < 4; ++gi) {
      const int n0 = gi * Hh + nh * 16;
      v8bf xi = *(const v8bf*)(xgs + (size_t)(n0 + l16) * Bb + r * 16 + lhi * 8);
      v8f acc;
#pragma unroll
      for (int v = 0; v < 8; ++v) acc[v] = bf2f(xi[v]);
#pragma unroll
      for (int kk = 0; kk < 4; ++kk) {
        const __bf16* wrow = whh_s + (size_t)(n0 + l16) * Hh + kk * 32 + lhi * 8;
        v8bf lo = *(const v8bf*)wrow;
        v8bf hi = *(const v8bf*)(wrow + 16);
        v16bf b;
#pragma unroll
        for (int q = 0; q < 8; ++q) { b[q] = lo[q]; b[q + 8] = hi[q]; }
        acc = __builtin_amdgcn_wmma_f32_16x16x32_bf16(false, afrag[kk], false, b,
                                                      (short)0, acc, false, false);
      }
#pragma unroll
      for (int v = 0; v < 8; ++v) gv[gi][v] = acc[v];
    }

    // --- pointwise gates, c/h update, writes (PyTorch gate order i,f,g,o)
    __bf16* hnew = hbuf + (p ^ 1) * Bb * Hh;
#pragma unroll
    for (int v = 0; v < 8; ++v) {
      const int m = r * 16 + v + lhi * 8;
      float ig = sigf(gv[0][v]);
      float fg = sigf(gv[1][v]);
      float gg = tanhf(gv[2][v]);
      float og = sigf(gv[3][v]);
      float c  = fg * cstate[v] + ig * gg;
      cstate[v] = c;
      float h  = og * tanhf(c);
      hnew[m * Hh + j] = f2bf(h);
      const size_t row = (size_t)tt * Bb + m;
      float xv = h;
      if (resid)  xv += resid[row * Hh + j];
      if (out_f)  out_f[row * out_stride + col_off + j] = xv;
      if (out_bf) out_bf[row * bf_stride + bf_off + j] = f2bf(xv);
      if (s == Tt - 1) {
        if (hn) hn[m * Hh + j] = h;
        if (cn) cn[m * Hh + j] = c;
      }
    }

#ifdef HAVE_ASYNC_LDS
    __builtin_amdgcn_s_wait_asynccnt(0);   // next slab resident in LDS
#endif
    __syncthreads();
    p ^= 1;
  }
}

// ---------------------------------------------------------------------------
extern "C" void kernel_launch(void* const* d_in, const int* in_sizes, int n_in,
                              void* d_out, int out_size, void* d_ws, size_t ws_size,
                              hipStream_t stream) {
  const int*   ids   = (const int*)  d_in[0];
  const float* emb   = (const float*)d_in[1];
  const float* w_ih0 = (const float*)d_in[2];   // [2,512,128]
  const float* w_hh0 = (const float*)d_in[3];   // [2,512,128]
  const float* b0    = (const float*)d_in[4];   // [2,512]
  const float* w_ih1 = (const float*)d_in[5];   // [512,256]
  const float* w_hh1 = (const float*)d_in[6];   // [512,128]
  const float* b1    = (const float*)d_in[7];   // [512]
  const float* w_ihr = (const float*)d_in[8];   // [6,512,128]
  const float* w_hhr = (const float*)d_in[9];   // [6,512,128]
  const float* br    = (const float*)d_in[10];  // [6,512]

  char* wsb = (char*)d_ws;
  size_t off = 0;
  auto alloc = [&](size_t bytes) {
    void* p = wsb + off;
    off = (off + bytes + 255) & ~(size_t)255;
    return p;
  };
  __bf16* xg    = (__bf16*)alloc((size_t)Tt * SLAB * 2);       // 32 MB, [T,4H,B]
  __bf16* wih0b = (__bf16*)alloc(2ull * G4 * Hh * 2);
  __bf16* whh0b = (__bf16*)alloc(2ull * G4 * Hh * 2);
  __bf16* wih1b = (__bf16*)alloc((size_t)G4 * 256 * 2);
  __bf16* whh1b = (__bf16*)alloc((size_t)G4 * Hh * 2);
  __bf16* wihrb = (__bf16*)alloc(6ull * G4 * Hh * 2);
  __bf16* whhrb = (__bf16*)alloc(6ull * G4 * Hh * 2);
  __bf16* x0bf  = (__bf16*)alloc((size_t)TBr * Hh * 2);        // 8 MB
  __bf16* catbf = (__bf16*)alloc((size_t)TBr * 2 * Hh * 2);    // 16 MB
  __bf16* xbfA  = (__bf16*)alloc((size_t)TBr * Hh * 2);
  __bf16* xbfB  = (__bf16*)alloc((size_t)TBr * Hh * 2);
  float*  xfA   = (float*) alloc((size_t)TBr * Hh * 4);        // 16 MB
  float*  xfB   = (float*) alloc((size_t)TBr * Hh * 4);

  auto cvt = [&](const float* src, __bf16* dst, int n) {
    cvt_bf16_kernel<<<(n + 255) / 256, 256, 0, stream>>>(src, dst, n);
  };
  cvt(w_ih0, wih0b, 2 * G4 * Hh);
  cvt(w_hh0, whh0b, 2 * G4 * Hh);
  cvt(w_ih1, wih1b, G4 * 256);
  cvt(w_hh1, whh1b, G4 * Hh);
  cvt(w_ihr, wihrb, 6 * G4 * Hh);
  cvt(w_hhr, whhrb, 6 * G4 * Hh);

  embed_kernel<<<(TBr * Hh + 255) / 256, 256, 0, stream>>>(ids, emb, x0bf, TBr * Hh);

  const size_t shmem = (size_t)G4 * Hh * 2      // whh      128 KB
                     + 2ull * Bb * Hh * 2       // h dbl     32 KB
                     + 2ull * SLAB * 2;         // xg dbl   128 KB  => 288 KB
  const int gemm_blocks = (TBr / 16) * (G4 / 16) / 8;             // 8192

  // ---- layer 0 forward / backward (outputs concat into catbf [T,B,2H])
  gemm_xg_kernel<<<gemm_blocks, 256, 0, stream>>>(x0bf, wih0b, b0, xg, Hh);
  lstm_scan_kernel<<<1, 1024, shmem, stream>>>(xg, whh0b,
      nullptr, 0, 0, catbf, 2 * Hh, 0, nullptr, nullptr, nullptr, 0);

  gemm_xg_kernel<<<gemm_blocks, 256, 0, stream>>>(x0bf, wih0b + (size_t)G4 * Hh,
                                                  b0 + G4, xg, Hh);
  lstm_scan_kernel<<<1, 1024, shmem, stream>>>(xg, whh0b + (size_t)G4 * Hh,
      nullptr, 0, 0, catbf, 2 * Hh, Hh, nullptr, nullptr, nullptr, 1);

  // ---- layer 1 (2H -> H)
  gemm_xg_kernel<<<gemm_blocks, 256, 0, stream>>>(catbf, wih1b, b1, xg, 2 * Hh);
  lstm_scan_kernel<<<1, 1024, shmem, stream>>>(xg, whh1b,
      xfA, Hh, 0, xbfA, Hh, 0, nullptr, nullptr, nullptr, 0);

  // ---- layers 2..7 (residual); last layer writes straight into d_out
  float* out_x  = (float*)d_out;
  float* out_hn = out_x + (size_t)TBr * Hh;
  float* out_cn = out_hn + (size_t)Bb * Hh;

  float*  prev_f  = xfA;
  __bf16* prev_bf = xbfA;
  for (int k = 0; k < 6; ++k) {
    gemm_xg_kernel<<<gemm_blocks, 256, 0, stream>>>(prev_bf,
        wihrb + (size_t)k * G4 * Hh, br + (size_t)k * G4, xg, Hh);
    const bool last = (k == 5);
    float*  of = last ? out_x   : ((k & 1) ? xfA  : xfB);
    __bf16* ob = last ? nullptr : ((k & 1) ? xbfA : xbfB);
    lstm_scan_kernel<<<1, 1024, shmem, stream>>>(xg,
        whhrb + (size_t)k * G4 * Hh, of, Hh, 0, ob, Hh, 0, prev_f,
        last ? out_hn : nullptr, last ? out_cn : nullptr, 0);
    prev_f  = of;
    prev_bf = ob;
  }
}
